// SpectralConv2d_16904991277632
// MI455X (gfx1250) — matrix operand order repaired and checked by reading, not compiled, not run on hardware
//
#include <hip/hip_runtime.h>
#include <hip/hip_bf16.h>

typedef __attribute__((ext_vector_type(16))) _Float16 v16h;
typedef __attribute__((ext_vector_type(8)))  float    v8f;

#define TWOPI 6.28318530717958647692f

// ---- workspace layout (bytes) ----
// tDFTW : [48][256]  f16  B^T for stage A1 (forward DFT along W)
// tEH   : [80][512]  f16  A   for stage A2 (forward DFT along H, complex via K-interleave)
// tE2   : [512][96]  f16  A   for stage C1 (inverse along H, 1/256 folded in)
// tDINV : [256][64]  f16  B^T for stage C2 (inverse along W, alpha/256 folded in)
// Xf    : [16][64][80][32] f16   (rows: re 0..39, im 40..79; cols kx 0..19 valid)
// Fout  : [16][64][80][32] f16   (rows: 2r+p;        cols kx 0..19 valid)
#define OFF_DFTW 0
#define OFF_EH   (48*256*2)
#define OFF_E2   (OFF_EH + 80*512*2)
#define OFF_DINV (OFF_E2 + 512*96*2)
#define OFF_XF   (OFF_DINV + 256*64*2)
#define OFF_FOUT (OFF_XF + 16*64*80*32*2)

union FragAB { v16h v; uint32_t u[8]; };
union FragC  { v8f  v; float    f[8]; };

// A fragment (16x32 f16), source row-major [M][K], ld in halfs (must be even).
// Lane layout: row = lane&15; elem e -> K = (e&7) + ((e>>3)<<4) + 8*(lane>>4)
__device__ __forceinline__ v16h ld_fragA(const _Float16* p, int ld, int row0, int k0) {
  int lane = threadIdx.x & 31;
  int m  = row0 + (lane & 15);
  int kb = k0 + ((lane >> 4) << 3);
  const uint32_t* q = (const uint32_t*)(p + m * ld);
  FragAB r;
  int j = kb >> 1;
#pragma unroll
  for (int pe = 0; pe < 4; ++pe) { r.u[pe] = q[j + pe]; r.u[pe + 4] = q[j + 8 + pe]; }
  return r.v;
}

// A fragment from an f32 source (convert on load).
__device__ __forceinline__ v16h ld_fragA_f32(const float* p, int ld, int row0, int k0) {
  int lane = threadIdx.x & 31;
  int m  = row0 + (lane & 15);
  int kb = k0 + ((lane >> 4) << 3);
  const float* q = p + m * ld + kb;
  v16h r;
#pragma unroll
  for (int e = 0; e < 8; ++e) r[e] = (_Float16)q[e];        // K = kb .. kb+7
#pragma unroll
  for (int e = 0; e < 8; ++e) r[8 + e] = (_Float16)q[16 + e]; // K = kb+16 .. kb+23
  return r;
}

// B fragment (32x16 f16) from TRANSPOSED storage [N][K], ldK in halfs (even).
// Lane layout: col = lane&15; elem e -> K = (e&7) + ((e>>3)<<4) + 8*(lane>>4)
__device__ __forceinline__ v16h ld_fragBT(const _Float16* pT, int ldK, int k0, int n0) {
  int lane = threadIdx.x & 31;
  int n  = n0 + (lane & 15);
  int kb = k0 + ((lane >> 4) << 3);
  const uint32_t* q = (const uint32_t*)(pT + n * ldK);
  FragAB r;
  int j = kb >> 1;
#pragma unroll
  for (int pe = 0; pe < 4; ++pe) { r.u[pe] = q[j + pe]; r.u[pe + 4] = q[j + 8 + pe]; }
  return r.v;
}

__device__ __forceinline__ v8f wmma_f16(v16h a, v16h b, v8f c) {
  return __builtin_amdgcn_wmma_f32_16x16x32_f16(false, a, false, b, (short)0, c, false, false);
}

// CDNA5 async global->LDS copy (ASYNCcnt-tracked, no VGPR round trip).
__device__ __forceinline__ void async_ld_b128(uint32_t lds_off, const float* gaddr) {
  asm volatile("global_load_async_to_lds_b128 %0, %1, off"
               :: "v"(lds_off), "v"(gaddr) : "memory");
}
__device__ __forceinline__ void wait_async0() {
  asm volatile("s_wait_asynccnt 0x0" ::: "memory");
}

// ---------------- twiddle-table generation (exact angle reduction mod 256) ----------------
__global__ void k_init_tables(_Float16* tDFTW, _Float16* tEH, _Float16* tE2, _Float16* tDINV) {
  const int total = 48*256 + 80*512 + 512*96 + 256*64;
  for (int idx = blockIdx.x * blockDim.x + threadIdx.x; idx < total;
       idx += gridDim.x * blockDim.x) {
    int t = idx;
    if (t < 48*256) {            // tDFTW^T [n][w]
      int n = t >> 8, w = t & 255;
      float v = 0.f;
      if (n < 20)      v =  cosf(TWOPI * (float)((w * n) & 255) * (1.f/256.f));
      else if (n < 40) v = -sinf(TWOPI * (float)((w * (n-20)) & 255) * (1.f/256.f));
      tDFTW[t] = (_Float16)v;
      continue;
    }
    t -= 48*256;
    if (t < 80*512) {            // tEH [r80][2h+p]
      int r80 = t >> 9, k = t & 511;
      int r  = (r80 < 40) ? r80 : r80 - 40;
      int ky = (r < 20) ? r : 216 + r;          // r>=20 -> ky = 236 + (r-20)
      int h = k >> 1, p = k & 1;
      float th = TWOPI * (float)((ky * h) & 255) * (1.f/256.f);
      float c = cosf(th), s = sinf(th);
      // X = sum_h e^{-i th} * T1 :  Xr rows (r80<40): [c, s] ; Xi rows: [-s, c]
      float v = (r80 < 40) ? (p ? s : c) : (p ? c : -s);
      tEH[t] = (_Float16)v;
      continue;
    }
    t -= 80*512;
    if (t < 512*96) {            // tE2 [part*256+h][2r+p]
      int m = t / 96, k = t % 96;
      int h = m & 255, part = m >> 8;
      float v = 0.f;
      if (k < 80) {
        int r = k >> 1, p = k & 1;
        int ky = (r < 20) ? r : 216 + r;
        float th = TWOPI * (float)((ky * h) & 255) * (1.f/256.f);
        float c = cosf(th), s = sinf(th);
        // S = sum_r e^{+i th} * F : Sre rows: [c, -s] ; Sim rows: [s, c]   (scaled 1/256)
        v = ((part == 0) ? (p ? -s : c) : (p ? c : s)) * (1.f/256.f);
      }
      tE2[t] = (_Float16)v;
      continue;
    }
    t -= 512*96;
    {                            // tDINV^T [w][2kx+p]
      int w = t >> 6, k = t & 63;
      int kx = k >> 1, p = k & 1;
      float v = 0.f;
      if (kx < 20) {
        float a  = (kx == 0) ? 1.f : 2.f;       // conjugate-symmetry doubling, DC once
        float th = TWOPI * (float)((kx * w) & 255) * (1.f/256.f);
        v = (p ? -a * sinf(th) : a * cosf(th)) * (1.f/256.f);
      }
      tDINV[t] = (_Float16)v;
    }
  }
}

// ---------------- forward: per (b,i) image, partial rfft2 on kept modes ----------------
__global__ void __launch_bounds__(256) k_forward(const float* __restrict__ x,
                                                 const _Float16* __restrict__ tDFTW,
                                                 const _Float16* __restrict__ tEH,
                                                 _Float16* __restrict__ Xf) {
  __shared__ float    sX[2][256 * 32];  // double-buffered x chunk [h][32 w] (f32, async-filled)
  __shared__ _Float16 sB2T[32 * 512];   // T1 as B^T for A2: [kx][2h+p]
  const int blk  = blockIdx.x;          // b*64 + i
  const int tid  = threadIdx.x;
  const int wave = tid >> 5;
  const float* xg = x + (long)blk * 256 * 256;

  { // zero sB2T (covers kx rows 20..31 padding and interleave target)
    uint32_t* z = (uint32_t*)sB2T;
    for (int i = tid; i < 32 * 512 / 2; i += 256) z[i] = 0u;
  }

  v8f zero = {0.f,0.f,0.f,0.f,0.f,0.f,0.f,0.f};
  v8f acc[6];
#pragma unroll
  for (int j = 0; j < 6; ++j) acc[j] = zero;

  // prime chunk 0 (each thread issues 8 b128 async copies)
#pragma unroll
  for (int it = 0; it < 8; ++it) {
    int idx = tid + it * 256;           // 0..2047
    int h = idx >> 3, wq = (idx & 7) << 2;
    async_ld_b128((uint32_t)(uintptr_t)(&sX[0][h * 32 + wq]), xg + h * 256 + wq);
  }

  // Stage A1: [256h x 256w] * [256w x 48] , K-chunked by 32, async double-buffered
  for (int kc = 0; kc < 8; ++kc) {
    wait_async0();                      // own async copies for chunk kc complete
    __syncthreads();                    // all waves' copies done; prev buffer fully consumed
    if (kc < 7) {
#pragma unroll
      for (int it = 0; it < 8; ++it) {
        int idx = tid + it * 256;
        int h = idx >> 3, wq = (idx & 7) << 2;
        async_ld_b128((uint32_t)(uintptr_t)(&sX[(kc + 1) & 1][h * 32 + wq]),
                      xg + h * 256 + (kc + 1) * 32 + wq);
      }
    }
#pragma unroll
    for (int j = 0; j < 6; ++j) {
      int t = wave + 8 * j;             // 48 tiles: mt in 0..15, nt in 0..2
      int mt = t / 3, nt = t % 3;
      v16h a = ld_fragA_f32(sX[kc & 1], 32, mt * 16, 0);
      v16h b = ld_fragBT(tDFTW, 256, kc * 32, nt * 16);
      acc[j] = wmma_f16(a, b, acc[j]);
    }
  }
  __syncthreads();
  { // scatter T1 -> sB2T  (col<20: re -> row 2h ; col 20..39: im -> row 2h+1)
    int lane = tid & 31, half_ = lane >> 4, nlo = lane & 15;
#pragma unroll
    for (int j = 0; j < 6; ++j) {
      int t = wave + 8 * j;
      int mt = t / 3, nt = t % 3;
      FragC c; c.v = acc[j];
#pragma unroll
      for (int e = 0; e < 8; ++e) {
        int h   = mt * 16 + e + 8 * half_;
        int col = nt * 16 + nlo;
        if (col < 40) {
          int p  = col >= 20;
          int kx = col - 20 * p;
          sB2T[kx * 512 + 2 * h + p] = (_Float16)c.f[e];
        }
      }
    }
  }
  __syncthreads();

  // Stage A2: [80 x 512] * [512 x 32] -> Xf
  for (int t = wave; t < 10; t += 8) {
    int mt = t >> 1, nt = t & 1;
    v8f a2 = zero;
    for (int kc = 0; kc < 16; ++kc) {
      v16h a = ld_fragA(tEH, 512, mt * 16, kc * 32);
      v16h b = ld_fragBT(sB2T, 512, kc * 32, nt * 16);
      a2 = wmma_f16(a, b, a2);
    }
    int lane = tid & 31, half_ = lane >> 4, nlo = lane & 15;
    FragC c; c.v = a2;
#pragma unroll
    for (int e = 0; e < 8; ++e) {
      int row = mt * 16 + e + 8 * half_;
      int col = nt * 16 + nlo;
      Xf[((long)blk * 80 + row) * 32 + col] = (_Float16)c.f[e];
    }
  }
}

// ---------------- mixing: per mode (r,kx), complex [16x64]x[64x64] ----------------
__global__ void __launch_bounds__(128) k_mix(const _Float16* __restrict__ Xf,
                                             const float* __restrict__ w1r,
                                             const float* __restrict__ w1i,
                                             const float* __restrict__ w2r,
                                             const float* __restrict__ w2i,
                                             _Float16* __restrict__ Fout) {
  __shared__ _Float16 sA[16 * 128];      // A: [b][2i+p]
  __shared__ _Float16 sBreT[64 * 128];   // B^T for real out: [o][2i+p]
  __shared__ _Float16 sBimT[64 * 128];   // B^T for imag out
  const int r  = blockIdx.x / 20;
  const int kx = blockIdx.x % 20;
  const int rr = (r < 20) ? r : r - 20;
  const float* wr = (r < 20) ? w1r : w2r;
  const float* wi = (r < 20) ? w1i : w2i;
  const int tid  = threadIdx.x;
  const int wave = tid >> 5;

  for (int idx = tid; idx < 16 * 128; idx += 128) {
    int b = idx >> 7, k = idx & 127;
    int i = k >> 1, p = k & 1;
    sA[idx] = Xf[(((long)b * 64 + i) * 80 + p * 40 + r) * 32 + kx];
  }
  for (int idx = tid; idx < 64 * 128; idx += 128) {
    int o = idx >> 7, k = idx & 127;
    int i = k >> 1, p = k & 1;
    long widx = (((long)i * 64 + o) * 20 + rr) * 20 + kx;
    float vr = wr[widx], vi = wi[widx];
    sBreT[o * 128 + k] = (_Float16)(p ? -vi : vr);   // Or = Xr*Wr - Xi*Wi
    sBimT[o * 128 + k] = (_Float16)(p ?  vr : vi);   // Oi = Xr*Wi + Xi*Wr
  }
  __syncthreads();

  v8f zero = {0.f,0.f,0.f,0.f,0.f,0.f,0.f,0.f};
  for (int t = wave; t < 8; t += 4) {
    int part = t >> 2, nt = t & 3;
    const _Float16* bT = part ? sBimT : sBreT;
    v8f acc = zero;
#pragma unroll
    for (int kc = 0; kc < 4; ++kc) {
      v16h a = ld_fragA(sA, 128, 0, kc * 32);
      v16h b = ld_fragBT(bT, 128, kc * 32, nt * 16);
      acc = wmma_f16(a, b, acc);
    }
    int lane = tid & 31, half_ = lane >> 4, nlo = lane & 15;
    FragC c; c.v = acc;
#pragma unroll
    for (int e = 0; e < 8; ++e) {
      int bb = e + 8 * half_;                  // batch
      int o  = nt * 16 + nlo;
      Fout[(((long)bb * 64 + o) * 80 + 2 * r + part) * 32 + kx] = (_Float16)c.f[e];
    }
  }
}

// ---------------- inverse: per (b,o) image, partial irfft2 ----------------
__global__ void __launch_bounds__(256) k_inverse(const _Float16* __restrict__ Fout,
                                                 const _Float16* __restrict__ tE2,
                                                 const _Float16* __restrict__ tDINV,
                                                 float* __restrict__ out) {
  __shared__ _Float16 sB3T[32 * 96];    // Fout as B^T: [kx][2r+p], padded zeros
  __shared__ _Float16 sS[256 * 64];     // S as A for C2: [h][2kx+p]
  const int blk  = blockIdx.x;          // b*64 + o
  const int tid  = threadIdx.x;
  const int wave = tid >> 5;
  const long fbase = (long)blk * 80 * 32;

  for (int idx = tid; idx < 32 * 96; idx += 256) {
    int n = idx / 96, k = idx % 96;
    _Float16 v = (_Float16)0.f;
    if (n < 20 && k < 80) v = Fout[fbase + (long)k * 32 + n];
    sB3T[idx] = v;
  }
  __syncthreads();

  v8f zero = {0.f,0.f,0.f,0.f,0.f,0.f,0.f,0.f};
  // Stage C1: [512 x 96] * [96 x 32] -> sS (scatter to [h][2kx+p])
#pragma unroll
  for (int j = 0; j < 8; ++j) {
    int t = wave + 8 * j;                // 64 tiles
    int mt = t >> 1, nt = t & 1;
    v8f acc = zero;
#pragma unroll
    for (int kc = 0; kc < 3; ++kc) {
      v16h a = ld_fragA(tE2, 96, mt * 16, kc * 32);
      v16h b = ld_fragBT(sB3T, 96, kc * 32, nt * 16);
      acc = wmma_f16(a, b, acc);
    }
    int lane = tid & 31, half_ = lane >> 4, nlo = lane & 15;
    FragC c; c.v = acc;
#pragma unroll
    for (int e = 0; e < 8; ++e) {
      int M = mt * 16 + e + 8 * half_;
      int h = M & 255, part = M >> 8;
      int kxv = nt * 16 + nlo;           // kx>=20 columns are exact zeros (B padded)
      sS[h * 64 + 2 * kxv + part] = (_Float16)c.f[e];
    }
  }
  __syncthreads();

  // Stage C2: [256 x 64] * [64 x 256] -> out
  for (int j = 0; j < 32; ++j) {
    int t = wave + 8 * j;                // 256 tiles
    int mt = t >> 4, nt = t & 15;
    v8f acc = zero;
#pragma unroll
    for (int kc = 0; kc < 2; ++kc) {
      v16h a = ld_fragA(sS, 64, mt * 16, kc * 32);
      v16h b = ld_fragBT(tDINV, 64, kc * 32, nt * 16);
      acc = wmma_f16(a, b, acc);
    }
    int lane = tid & 31, half_ = lane >> 4, nlo = lane & 15;
    FragC c; c.v = acc;
#pragma unroll
    for (int e = 0; e < 8; ++e) {
      int h = mt * 16 + e + 8 * half_;
      int w = nt * 16 + nlo;
      out[((long)blk * 256 + h) * 256 + w] = c.f[e];
    }
  }
}

extern "C" void kernel_launch(void* const* d_in, const int* in_sizes, int n_in,
                              void* d_out, int out_size, void* d_ws, size_t ws_size,
                              hipStream_t stream) {
  const float* x   = (const float*)d_in[0];
  const float* w1r = (const float*)d_in[1];
  const float* w1i = (const float*)d_in[2];
  const float* w2r = (const float*)d_in[3];
  const float* w2i = (const float*)d_in[4];
  char* ws = (char*)d_ws;
  _Float16* tDFTW = (_Float16*)(ws + OFF_DFTW);
  _Float16* tEH   = (_Float16*)(ws + OFF_EH);
  _Float16* tE2   = (_Float16*)(ws + OFF_E2);
  _Float16* tDINV = (_Float16*)(ws + OFF_DINV);
  _Float16* Xf    = (_Float16*)(ws + OFF_XF);
  _Float16* Fout  = (_Float16*)(ws + OFF_FOUT);

  k_init_tables<<<128, 256, 0, stream>>>(tDFTW, tEH, tE2, tDINV);
  k_forward   <<<16 * 64, 256, 0, stream>>>(x, tDFTW, tEH, Xf);
  k_mix       <<<40 * 20, 128, 0, stream>>>(Xf, w1r, w1i, w2r, w2i, Fout);
  k_inverse   <<<16 * 64, 256, 0, stream>>>(Fout, tE2, tDINV, (float*)d_out);
}
